// LocalClosedFormResidualHead_40200893891290
// MI455X (gfx1250) — matrix operand order repaired and checked by reading, not compiled, not run on hardware
//
#include <hip/hip_runtime.h>
#include <math.h>

#define B_SZ 32768
#define C_SZ 100
#define K_SZ 8
#define D_SZ 512
#define NTILE 57                 /* ceil(900/16) column tiles (800 protos + 100 W cols) */
#define NKB 16                   /* 512 / 32 k-blocks */
#define EPSN 1e-8f
#define RIDGE_C 0.1f
#define WS_W (0.5f / 8.0f)       /* same-class support weight  */
#define WO_W (0.5f / 99.0f)      /* opposing-centers weight    */
#define INV_SUB_T 20.0f          /* 1/0.05 */
#define INV_PRIOR_T (1.0f / 0.07f)
#define INV_ROUT_T (1.0f / 0.07f)
#define CG_ITERS 30

typedef __attribute__((ext_vector_type(16))) __bf16 v16bf;
typedef __attribute__((ext_vector_type(8))) float v8f;

/* split two f32 into bf16 hi/lo planes, packed as (lo16=elem0, hi16=elem1) */
__device__ __forceinline__ void split_pack(float x0, float x1, unsigned& hiw, unsigned& low) {
  __bf16 h0 = (__bf16)x0, h1 = (__bf16)x1;
  __bf16 l0 = (__bf16)(x0 - (float)h0), l1 = (__bf16)(x1 - (float)h1);
  hiw = ((unsigned)__builtin_bit_cast(unsigned short, h1) << 16) |
        (unsigned)__builtin_bit_cast(unsigned short, h0);
  low = ((unsigned)__builtin_bit_cast(unsigned short, l1) << 16) |
        (unsigned)__builtin_bit_cast(unsigned short, l0);
}

/* k-offset of element pair qh inside a 32-k block, ISA 16-bit A 16x32 layout */
__device__ __forceinline__ int a_kin(int qh, int lane) {
  return (qh < 4) ? (((lane < 16) ? 0 : 8) + 2 * qh)
                  : (((lane < 16) ? 16 : 24) + 2 * (qh - 4));
}
/* k-offset of element pair qh inside a 32-k block, same k order as a_kin */
__device__ __forceinline__ int b_kin(int qh, int lane) {
  return (qh < 4) ? (((lane < 16) ? 0 : 8) + 2 * qh)
                  : (((lane < 16) ? 16 : 24) + 2 * (qh - 4));
}

__device__ __forceinline__ float block_reduce_sum(float v, float* sh, int tid, int nthr) {
#pragma unroll
  for (int m = 16; m > 0; m >>= 1) v += __shfl_xor(v, m, 32);
  int wid = tid >> 5;
  if ((tid & 31) == 0) sh[wid] = v;
  __syncthreads();
  int nw = nthr >> 5;
  float r = (tid < nw) ? sh[tid] : 0.0f;
  if (tid < 32) {
#pragma unroll
    for (int m = 16; m > 0; m >>= 1) r += __shfl_xor(r, m, 32);
    if (tid == 0) sh[0] = r;
  }
  __syncthreads();
  float out = sh[0];
  __syncthreads();
  return out;
}

/* ---- row-wise L2 normalize (rows of length 512, 256 threads/row) ---- */
__global__ void k_l2norm_rows(const float* __restrict__ in, float* __restrict__ out) {
  __shared__ float sh[16];
  int row = blockIdx.x, tid = threadIdx.x;
  const float* ip = in + (size_t)row * D_SZ;
  float v0 = ip[tid], v1 = ip[tid + 256];
  float s = block_reduce_sum(v0 * v0 + v1 * v1, sh, tid, 256);
  float inv = 1.0f / fmaxf(sqrtf(s), EPSN);
  float* op = out + (size_t)row * D_SZ;
  op[tid] = v0 * inv;
  op[tid + 256] = v1 * inv;
}

/* ---- class centers: l2norm(mean_k Pn[c,k,:]) ---- */
__global__ void k_centers(const float* __restrict__ Pn, float* __restrict__ centers) {
  __shared__ float sh[16];
  int c = blockIdx.x, tid = threadIdx.x;
  float m0 = 0.0f, m1 = 0.0f;
  for (int k = 0; k < K_SZ; ++k) {
    const float* p = Pn + (size_t)(c * K_SZ + k) * D_SZ;
    m0 += p[tid];
    m1 += p[tid + 256];
  }
  m0 *= (1.0f / K_SZ);
  m1 *= (1.0f / K_SZ);
  float s = block_reduce_sum(m0 * m0 + m1 * m1, sh, tid, 256);
  float inv = 1.0f / fmaxf(sqrtf(s), EPSN);
  centers[c * D_SZ + tid] = m0 * inv;
  centers[c * D_SZ + tid + 256] = m1 * inv;
}

__global__ void k_csum(const float* __restrict__ centers, float* __restrict__ csum) {
  int tid = threadIdx.x;
  float s = 0.0f;
  for (int c = 0; c < C_SZ; ++c) s += centers[c * D_SZ + tid];
  csum[tid] = s;
}

/* ---- per-class matrix-free CG ridge solve: A_c x = b_c, one block per class ---- */
__global__ void __launch_bounds__(512) k_cg_solve(const float* __restrict__ Pn,
                                                  const float* __restrict__ centers,
                                                  const float* __restrict__ csum,
                                                  float* __restrict__ Wm) {
  __shared__ float sP[K_SZ * D_SZ];
  __shared__ float sp_vec[D_SZ];
  __shared__ float sv[C_SZ];
  __shared__ float su[K_SZ];
  __shared__ float sred[16];
  int c = blockIdx.x, tid = threadIdx.x;
  for (int f = tid; f < K_SZ * D_SZ; f += 512) sP[f] = Pn[(size_t)c * K_SZ * D_SZ + f];
  float cc = centers[c * D_SZ + tid];
  float cs = csum[tid];
  __syncthreads();

  float b = 0.0f;
#pragma unroll
  for (int k = 0; k < K_SZ; ++k) b += sP[k * D_SZ + tid];
  b = WS_W * b - WO_W * (cs - cc);

  float x = 0.0f, r = b;
  sp_vec[tid] = b;
  __syncthreads();
  float rs = block_reduce_sum(r * r, sred, tid, 512);

  int wave = tid >> 5, lane = tid & 31;
  for (int it = 0; it < CG_ITERS; ++it) {
    if (wave < K_SZ) {
      float acc = 0.0f;
      const float* pr = sP + wave * D_SZ;
      for (int e = lane; e < D_SZ; e += 32) acc += pr[e] * sp_vec[e];
#pragma unroll
      for (int m = 16; m > 0; m >>= 1) acc += __shfl_xor(acc, m, 32);
      if (lane == 0) su[wave] = acc;
    }
    for (int row = wave; row < C_SZ; row += 16) {
      float acc = 0.0f;
      const float* cr = centers + row * D_SZ;
      for (int e = lane; e < D_SZ; e += 32) acc += cr[e] * sp_vec[e];
#pragma unroll
      for (int m = 16; m > 0; m >>= 1) acc += __shfl_xor(acc, m, 32);
      if (lane == 0) sv[row] = acc;
    }
    __syncthreads();
    float y = 0.0f;
    for (int row = 0; row < C_SZ; ++row) y += centers[row * D_SZ + tid] * sv[row];
    float dcp = block_reduce_sum(cc * sp_vec[tid], sred, tid, 512);
    float pu = 0.0f;
#pragma unroll
    for (int k = 0; k < K_SZ; ++k) pu += su[k] * sP[k * D_SZ + tid];
    float q = RIDGE_C * sp_vec[tid] + WS_W * pu + WO_W * (y - cc * dcp);

    float pq = block_reduce_sum(sp_vec[tid] * q, sred, tid, 512);
    float alpha = rs / fmaxf(pq, 1e-30f);
    x += alpha * sp_vec[tid];
    r -= alpha * q;
    float rs2 = block_reduce_sum(r * r, sred, tid, 512);
    float beta = rs2 / fmaxf(rs, 1e-30f);
    rs = rs2;
    float pnew = r + beta * sp_vec[tid];
    __syncthreads();
    sp_vec[tid] = pnew;
    __syncthreads();
  }
  Wm[c * D_SZ + tid] = x;
}

/* ---- pre-swizzle combined [Pn;W;0] matrix into bf16 hi/lo B-fragment planes ----
   layout per tile: [kb 0..15][plane hi/lo][lane 0..31][16 bf16 in VGPR order]      */
__global__ void k_bfrag(const float* __restrict__ Pn, const float* __restrict__ Wm,
                        __bf16* __restrict__ Bf) {
  int u = blockIdx.x * 256 + threadIdx.x; /* 912*256 = 57*4096 units */
  int qh = u & 7, L = (u >> 3) & 31, i = (u >> 8) & 15, t = u >> 12;
  int k = i * 32 + b_kin(qh, L);
  int n = t * 16 + (L & 15);
  float2 v;
  v.x = 0.0f;
  v.y = 0.0f;
  if (n < 800)
    v = *reinterpret_cast<const float2*>(Pn + (size_t)n * D_SZ + k);
  else if (n < 900)
    v = *reinterpret_cast<const float2*>(Wm + (size_t)(n - 800) * D_SZ + k);
  unsigned hiw, low;
  split_pack(v.x, v.y, hiw, low);
  unsigned* B32 = reinterpret_cast<unsigned*>(Bf);
  size_t base = (size_t)t * 8192 + (size_t)i * 512 + L * 8 + qh;
  B32[base] = hiw;        /* hi plane */
  B32[base + 256] = low;  /* lo plane */
}

/* ---- epilogue: softmax-pool 2 classes per proto tile / residual add for W tiles ---- */
__device__ __forceinline__ void epilogue_tile(int t, int lane, v8f s0, v8f s1, float* sOut) {
  int rhalf = (lane >> 4) * 8;
  if (t < 50) {
    int cls = (t * 16 + (lane & 15)) >> 3;
#pragma unroll
    for (int j = 0; j < 8; ++j) {
      float p0 = s0[j], p1 = s1[j];
      float e0 = __expf(p0 * INV_SUB_T), n0 = e0 * p0;
      float e1 = __expf(p1 * INV_SUB_T), n1 = e1 * p1;
#pragma unroll
      for (int m = 1; m <= 4; m <<= 1) {
        e0 += __shfl_xor(e0, m, 32);
        n0 += __shfl_xor(n0, m, 32);
        e1 += __shfl_xor(e1, m, 32);
        n1 += __shfl_xor(n1, m, 32);
      }
      if ((lane & 7) == 0) {
        atomicAdd(&sOut[(rhalf + j) * C_SZ + cls], (n0 / e0) * INV_PRIOR_T);
        atomicAdd(&sOut[(16 + rhalf + j) * C_SZ + cls], (n1 / e1) * INV_PRIOR_T);
      }
    }
  } else {
    int cW = t * 16 + (lane & 15) - 800;
    if (cW < C_SZ) {
#pragma unroll
      for (int j = 0; j < 8; ++j) {
        atomicAdd(&sOut[(rhalf + j) * C_SZ + cW], s0[j] * INV_ROUT_T);
        atomicAdd(&sOut[(16 + rhalf + j) * C_SZ + cW], s1[j] * INV_ROUT_T);
      }
    }
  }
}

/* ---- fused split-bf16 WMMA GEMM + softmax-pool + prior/residual combine ----
   M=32 rows/workgroup (2 strips), 2 column tiles per wave sharing A-frag loads. */
__global__ void __launch_bounds__(256) k_fused_gemm(const float* __restrict__ hn,
                                                    const __bf16* __restrict__ Bf,
                                                    float* __restrict__ out) {
  __shared__ __align__(32) __bf16 sA[32768]; /* 64KB: 2 strips x 16kb x hi/lo x 32 lanes x 16 */
  __shared__ float sOut[32 * C_SZ];          /* 12.8 KB */
  int tid = threadIdx.x, blk = blockIdx.x;

  for (int f = tid; f < 32 * C_SZ; f += 256) sOut[f] = 0.0f;
  unsigned* sA32 = reinterpret_cast<unsigned*>(sA);
  for (int u = tid; u < 8192; u += 256) {
    int qh = u & 7, L = (u >> 3) & 31, i = (u >> 8) & 15, st = u >> 12;
    int k = i * 32 + a_kin(qh, L);
    int row = blk * 32 + st * 16 + (L & 15);
    const float2 v = *reinterpret_cast<const float2*>(hn + (size_t)row * D_SZ + k);
    unsigned hiw, low;
    split_pack(v.x, v.y, hiw, low);
    int base = st * 8192 + i * 512 + L * 8 + qh;
    sA32[base] = hiw;
    sA32[base + 256] = low;
  }
  __syncthreads();

  int wave = tid >> 5, lane = tid & 31;
  const v16bf* aB = reinterpret_cast<const v16bf*>(sA) + lane;

  for (int t0 = wave * 2; t0 < NTILE; t0 += 16) {
    int t1 = t0 + 1;
    bool has1 = (t1 < NTILE);
    int t1c = has1 ? t1 : t0;
    const v16bf* b0 = reinterpret_cast<const v16bf*>(Bf) + (size_t)t0 * 1024 + lane;
    const v16bf* b1 = reinterpret_cast<const v16bf*>(Bf) + (size_t)t1c * 1024 + lane;
    v8f a00 = {0, 0, 0, 0, 0, 0, 0, 0};
    v8f a10 = a00, a01 = a00, a11 = a00;
    for (int i = 0; i < NKB; ++i) {
      v16bf Bh0 = b0[i * 64], Bl0 = b0[i * 64 + 32];
      v16bf Bh1 = b1[i * 64], Bl1 = b1[i * 64 + 32];
      v16bf Ah0 = aB[i * 64], Al0 = aB[i * 64 + 32];
      v16bf Ah1 = aB[1024 + i * 64], Al1 = aB[1024 + i * 64 + 32];
      /* hi*hi */
      a00 = __builtin_amdgcn_wmma_f32_16x16x32_bf16(false, Ah0, false, Bh0, (short)0, a00, false, false);
      a10 = __builtin_amdgcn_wmma_f32_16x16x32_bf16(false, Ah1, false, Bh0, (short)0, a10, false, false);
      a01 = __builtin_amdgcn_wmma_f32_16x16x32_bf16(false, Ah0, false, Bh1, (short)0, a01, false, false);
      a11 = __builtin_amdgcn_wmma_f32_16x16x32_bf16(false, Ah1, false, Bh1, (short)0, a11, false, false);
      /* hi*lo */
      a00 = __builtin_amdgcn_wmma_f32_16x16x32_bf16(false, Ah0, false, Bl0, (short)0, a00, false, false);
      a10 = __builtin_amdgcn_wmma_f32_16x16x32_bf16(false, Ah1, false, Bl0, (short)0, a10, false, false);
      a01 = __builtin_amdgcn_wmma_f32_16x16x32_bf16(false, Ah0, false, Bl1, (short)0, a01, false, false);
      a11 = __builtin_amdgcn_wmma_f32_16x16x32_bf16(false, Ah1, false, Bl1, (short)0, a11, false, false);
      /* lo*hi */
      a00 = __builtin_amdgcn_wmma_f32_16x16x32_bf16(false, Al0, false, Bh0, (short)0, a00, false, false);
      a10 = __builtin_amdgcn_wmma_f32_16x16x32_bf16(false, Al1, false, Bh0, (short)0, a10, false, false);
      a01 = __builtin_amdgcn_wmma_f32_16x16x32_bf16(false, Al0, false, Bh1, (short)0, a01, false, false);
      a11 = __builtin_amdgcn_wmma_f32_16x16x32_bf16(false, Al1, false, Bh1, (short)0, a11, false, false);
    }
    epilogue_tile(t0, lane, a00, a10, sOut);
    if (has1) epilogue_tile(t1, lane, a01, a11, sOut);
  }
  __syncthreads();
  for (int f = tid; f < 32 * C_SZ; f += 256) {
    int r = f / C_SZ, c = f % C_SZ;
    out[((size_t)blk * 32 + r) * C_SZ + c] = sOut[f];
  }
}

extern "C" void kernel_launch(void* const* d_in, const int* in_sizes, int n_in,
                              void* d_out, int out_size, void* d_ws, size_t ws_size,
                              hipStream_t stream) {
  (void)in_sizes; (void)n_in; (void)out_size; (void)ws_size;
  const float* h = (const float*)d_in[0];
  const float* protos = (const float*)d_in[1];
  float* out = (float*)d_out;

  float* ws = (float*)d_ws;
  float* hn = ws;                                   /* 32768*512 f32              */
  float* Pn = hn + (size_t)B_SZ * D_SZ;             /* 800*512 f32                */
  float* centers = Pn + (size_t)C_SZ * K_SZ * D_SZ; /* 100*512 f32                */
  float* csum = centers + (size_t)C_SZ * D_SZ;      /* 512 f32                    */
  float* Wm = csum + D_SZ;                          /* 100*512 f32                */
  __bf16* Bf = (__bf16*)(Wm + (size_t)C_SZ * D_SZ); /* 57*16384 bf16 (1.87 MB)    */

  k_l2norm_rows<<<B_SZ, 256, 0, stream>>>(h, hn);
  k_l2norm_rows<<<C_SZ * K_SZ, 256, 0, stream>>>(protos, Pn);
  k_centers<<<C_SZ, 256, 0, stream>>>(Pn, centers);
  k_csum<<<1, 512, 0, stream>>>(centers, csum);
  k_cg_solve<<<C_SZ, 512, 0, stream>>>(Pn, centers, csum, Wm);
  k_bfrag<<<912, 256, 0, stream>>>(Pn, Wm, Bf);
  k_fused_gemm<<<B_SZ / 32, 256, 0, stream>>>(hn, Bf, out);
}
